// coRNN_19155554140910
// MI455X (gfx1250) — matrix operand order repaired and checked by reading, not compiled, not run on hardware
//
#include <hip/hip_runtime.h>
#include <hip/hip_bf16.h>
#include <stdint.h>

typedef __attribute__((ext_vector_type(16))) _Float16 v16h;
typedef __attribute__((ext_vector_type(8)))  float    v8f;
typedef __attribute__((ext_vector_type(4)))  int      v4i;

typedef __attribute__((address_space(1))) v4i g_v4i;   // global
typedef __attribute__((address_space(3))) v4i l_v4i;   // LDS

#define T_STEPS 512
#define BATCH   256
#define N_INP   128
#define N_HID   512
#define N_OUT   10
#define FAN_IN  (N_INP + 2 * N_HID)   // 1152
#define DT      0.042f
#define GAMMA   2.7f
#define EPSILON 4.7f

#if __has_builtin(__builtin_amdgcn_tanhf)
#define TANHF(v) __builtin_amdgcn_tanhf(v)
#else
#define TANHF(v) tanhf(v)
#endif

#define HAS_ASYNC_LDS __has_builtin(__builtin_amdgcn_global_load_async_to_lds_b128)

__device__ __forceinline__ _Float16 to_h(float f) { return (_Float16)f; }

// ---------------------------------------------------------------------------
// One-time init: W^T -> f16, zero hy/hz, build act0 = [f16(x_0) | 0 | 0]
// ---------------------------------------------------------------------------
__global__ void corn_init_kernel(const float* __restrict__ x,
                                 const float* __restrict__ W,
                                 _Float16* __restrict__ Wt,      // [N_HID][FAN_IN]
                                 _Float16* __restrict__ act0,    // [BATCH][FAN_IN]
                                 float* __restrict__ hy,
                                 float* __restrict__ hz) {
  int idx    = blockIdx.x * blockDim.x + threadIdx.x;
  int stride = gridDim.x * blockDim.x;

  for (int i = idx; i < FAN_IN * N_HID; i += stride) {
    int n = i / FAN_IN;
    int k = i - n * FAN_IN;
    Wt[i] = to_h(W[k * N_HID + n]);     // Wt[n][k] = W[k][n]
  }
  for (int i = idx; i < BATCH * N_HID; i += stride) {
    hy[i] = 0.0f;
    hz[i] = 0.0f;
  }
  for (int i = idx; i < BATCH * FAN_IN; i += stride) {
    int row = i / FAN_IN;
    int col = i - row * FAN_IN;
    _Float16 v = (_Float16)0.0f;
    if (col < N_INP) v = to_h(x[row * N_INP + col]);
    act0[i] = v;
  }
}

// ---------------------------------------------------------------------------
// One recurrence step:
//   pre = tanh(act_cur @ W + b)      (WMMA f16 -> f32, 512 tiles of 16x16)
//   hz += DT*(pre - GAMMA*hy - EPSILON*hz); hy += DT*hz   (fused epilogue)
//   act_next <- [f16(x[t+1]) | f16(hz) | f16(hy)]          (double-buffered)
// Grid: 64 blocks x 256 threads (8 waves). All 8 waves of a block share one
// n_tile, so the 16-column W slice (36 KB) is staged once into LDS via the
// gfx1250 async-to-LDS path (ASYNCcnt), then read back as B fragments.
// ---------------------------------------------------------------------------
__global__ void __launch_bounds__(256) corn_step_kernel(
    const float*    __restrict__ x,        // [T][BATCH][N_INP]
    const _Float16* __restrict__ Wt,       // [N_HID][FAN_IN]
    const float*    __restrict__ bias,     // [N_HID]
    const _Float16* __restrict__ act_cur,  // [BATCH][FAN_IN]
    _Float16*       __restrict__ act_next, // [BATCH][FAN_IN]
    float* __restrict__ hy,
    float* __restrict__ hz,
    int t) {
  __shared__ _Float16 Bs[16 * FAN_IN];     // 36 KB: W columns for this block

  // Part 1: stage x[t+1] into act_next's x-section (disjoint from epilogue's
  // z/y sections; act_cur != act_next so no read/write hazard). Prefetch the
  // x[t+2] slab into cache to hide HBM latency across the sequential chain.
  if (t + 1 < T_STEPS) {
    const float* xt = x + (size_t)(t + 1) * BATCH * N_INP;
    int tid = blockIdx.x * blockDim.x + threadIdx.x;
    for (int i = tid; i < BATCH * N_INP; i += gridDim.x * blockDim.x) {
      int row = i / N_INP;
      int col = i - row * N_INP;
      act_next[row * FAN_IN + col] = to_h(xt[i]);
      if (t + 2 < T_STEPS)
        __builtin_prefetch(xt + (size_t)BATCH * N_INP + i, 0, 1);
    }
  }

  // Part 2a: stage the block-shared B slice (16 cols x 1152 K, f16) into LDS.
  {
    const int ntb = blockIdx.x >> 1;                       // block's n_tile
    const _Float16* src = Wt + (size_t)ntb * 16 * FAN_IN;  // 2304 uint4s
    const int tid = threadIdx.x;
#if HAS_ASYNC_LDS
#pragma unroll
    for (int j = 0; j < 9; ++j) {
      int i = tid + 256 * j;                               // uint4 index
      __builtin_amdgcn_global_load_async_to_lds_b128(
          (g_v4i*)(src + i * 8),
          (l_v4i*)(Bs + i * 8),
          /*offset=*/0, /*cpol=*/0);
    }
#if __has_builtin(__builtin_amdgcn_s_wait_asynccnt)
    __builtin_amdgcn_s_wait_asynccnt(0);
#else
    asm volatile("s_wait_asynccnt 0x0" ::: "memory");
#endif
#else
    const uint4* s4 = (const uint4*)src;
    uint4* d4 = (uint4*)Bs;
#pragma unroll
    for (int j = 0; j < 9; ++j) {
      int i = tid + 256 * j;
      d4[i] = s4[i];
    }
#endif
    __syncthreads();
  }

  // Part 2b: WMMA GEMM tile (16x16 output per wave).
  const int wave   = threadIdx.x >> 5;
  const int lane   = threadIdx.x & 31;
  const int tile   = blockIdx.x * 8 + wave;  // 0..511
  const int n_tile = tile >> 4;              // == blockIdx.x >> 1
  const int m_tile = tile & 15;              // 0..15
  const int half   = lane >> 4;              // 0 or 1
  const int lm     = lane & 15;
  const int n      = n_tile * 16 + lm;       // output column (hidden unit)
  const int am     = m_tile * 16 + lm;       // A-fragment row (batch)

  const _Float16* arow = act_cur + am * FAN_IN;
  const _Float16* brow = Bs + lm * FAN_IN;   // B column from LDS

  v8f c = {};
  for (int k0 = 0; k0 < FAN_IN; k0 += 32) {
    union { v16h v; uint4 q[2]; } a, bf;
    // A 16x32 f16: elems 0-7 = K[half*8 .. +7], elems 8-15 = K[16+half*8 .. +7]
    a.q[0] = *reinterpret_cast<const uint4*>(arow + k0 + half * 8);
    a.q[1] = *reinterpret_cast<const uint4*>(arow + k0 + 16 + half * 8);
    // B 32x16 f16 (W pre-transposed): elems 0-7 = K[half*8 .. +7], elems 8-15 = K[16+half*8 .. +7] (LDS)
    bf.q[0] = *reinterpret_cast<const uint4*>(brow + k0 + half * 8);
    bf.q[1] = *reinterpret_cast<const uint4*>(brow + k0 + 16 + half * 8);
    c = __builtin_amdgcn_wmma_f32_16x16x32_f16(
        /*neg_a=*/false, a.v, /*neg_b=*/false, bf.v,
        /*c_mod=*/(short)0, c, /*reuse_a=*/false, /*reuse_b=*/false);
  }

  // Part 3: fused coRNN update epilogue (each lane owns 8 C elements).
  const float bn = bias[n];
#pragma unroll
  for (int r = 0; r < 8; ++r) {
    int mm  = m_tile * 16 + half * 8 + r;   // C/D layout row
    int idx = mm * N_HID + n;
    float pre = TANHF(c[r] + bn);
    float z = hz[idx];
    float y = hy[idx];
    float zn = z + DT * (pre - GAMMA * y - EPSILON * z);
    float yn = y + DT * zn;
    hz[idx] = zn;
    hy[idx] = yn;
    act_next[mm * FAN_IN + N_INP + n]         = to_h(zn);
    act_next[mm * FAN_IN + N_INP + N_HID + n] = to_h(yn);
  }
}

// ---------------------------------------------------------------------------
// out = hy @ Wout + bout   ([256x512]@[512x10] — tiny, scalar fp32)
// ---------------------------------------------------------------------------
__global__ void corn_out_kernel(const float* __restrict__ hy,
                                const float* __restrict__ Wout,
                                const float* __restrict__ bout,
                                float* __restrict__ out) {
  int idx = blockIdx.x * blockDim.x + threadIdx.x;
  if (idx >= BATCH * N_OUT) return;
  int m = idx / N_OUT;
  int o = idx - m * N_OUT;
  float acc = bout[o];
  const float* hrow = hy + m * N_HID;
  for (int k = 0; k < N_HID; ++k) acc += hrow[k] * Wout[k * N_OUT + o];
  out[idx] = acc;
}

// ---------------------------------------------------------------------------
extern "C" void kernel_launch(void* const* d_in, const int* in_sizes, int n_in,
                              void* d_out, int out_size, void* d_ws, size_t ws_size,
                              hipStream_t stream) {
  (void)in_sizes; (void)n_in; (void)out_size; (void)ws_size;

  const float* x    = (const float*)d_in[0];  // [T,B,N_INP]
  const float* W    = (const float*)d_in[1];  // [FAN_IN, N_HID]
  const float* b    = (const float*)d_in[2];  // [N_HID]
  const float* Wout = (const float*)d_in[3];  // [N_HID, N_OUT]
  const float* bout = (const float*)d_in[4];  // [N_OUT]

  char* ws = (char*)d_ws;
  size_t off = 0;
  auto carve = [&](size_t bytes) -> char* {
    char* p = ws + off;
    off += (bytes + 255) & ~size_t(255);
    return p;
  };
  _Float16* Wt   = (_Float16*)carve((size_t)FAN_IN * N_HID * sizeof(_Float16));
  _Float16* act0 = (_Float16*)carve((size_t)BATCH * FAN_IN * sizeof(_Float16));
  _Float16* act1 = (_Float16*)carve((size_t)BATCH * FAN_IN * sizeof(_Float16));
  float*    hy   = (float*)carve((size_t)BATCH * N_HID * sizeof(float));
  float*    hz   = (float*)carve((size_t)BATCH * N_HID * sizeof(float));

  corn_init_kernel<<<512, 256, 0, stream>>>(x, W, Wt, act0, hy, hz);

  _Float16* cur = act0;
  _Float16* nxt = act1;
  for (int t = 0; t < T_STEPS; ++t) {
    corn_step_kernel<<<64, 256, 0, stream>>>(x, Wt, b, cur, nxt, hy, hz, t);
    _Float16* tmp = cur; cur = nxt; nxt = tmp;
  }

  corn_out_kernel<<<(BATCH * N_OUT + 255) / 256, 256, 0, stream>>>(
      hy, Wout, bout, (float*)d_out);
}
